// NodeEmbedding_85057532330251
// MI455X (gfx1250) — matrix operand order repaired and checked by reading, not compiled, not run on hardware
//
#include <hip/hip_runtime.h>
#include <hip/hip_bf16.h>

// GGNN propagation, B=32, E=4 (2E=8 with transposes), N=512, D=256, 8 steps.
// All GEMMs via V_WMMA_F32_16X16X32_BF16. adj converted once to bf16 (134MB,
// resident in the 192MB L2 across all 8 steps). Round 3: explicit double-
// buffered operand pipelines so WMMAs overlap loads (no s_wait_loadcnt 0x0
// fences from register recycling).

#define Bb 32
#define EE 8          // 2*E
#define NN 512
#define DD 256
#define STEPS 8

typedef __attribute__((ext_vector_type(16))) __bf16 v16bf;
typedef __attribute__((ext_vector_type(8)))  __bf16 v8bf;
typedef __attribute__((ext_vector_type(8)))  float  v8f;

// ---- WMMA helpers -----------------------------------------------------------

__device__ __forceinline__ v8f wmma_bf16(v16bf a, v16bf b, v8f c) {
  // D = A(16x32 bf16) * B(32x16 bf16) + C(16x16 f32)
  return __builtin_amdgcn_wmma_f32_16x16x32_bf16(
      /*neg_a=*/false, a, /*neg_b=*/false, b,
      /*c_mod=*/(short)0, c, /*reuse_a=*/false, /*reuse_b=*/false);
}

// A-fragment (16x32 bf16) from a row-major bf16 matrix.
// ISA layout: lanes 0-15 -> M=lane, halves {K0..7, K16..23};
//             lanes 16-31 -> M=lane-16, halves {K8..15, K24..31}.
__device__ __forceinline__ v16bf load_a_frag(const __bf16* A, int lda,
                                             int row_base, int kk, int lane) {
  int r  = lane & 15;
  int ko = kk + ((lane & 16) ? 8 : 0);
  const __bf16* p = A + (long)(row_base + r) * lda + ko;
  v8bf lo = *(const v8bf*)(p);        // K = ko .. ko+7
  v8bf hi = *(const v8bf*)(p + 16);   // K = ko+16 .. ko+23
  return __builtin_shufflevector(lo, hi, 0,1,2,3,4,5,6,7,8,9,10,11,12,13,14,15);
}

// B-fragments are pre-swizzled: fragment f=(kt*ntiles+nt) is 32 lanes x 16 bf16
// contiguous (1KB), so the load is one v16bf (two b128) per lane.
__device__ __forceinline__ v16bf load_b_frag(const __bf16* sw, int frag, int lane) {
  return ((const v16bf*)(sw + ((long)frag << 9)))[lane];
}

// ---- one-time preprocessing -------------------------------------------------

// adj_full bf16: e<4 copy, e>=4 transpose; fused per-row sums for the bias fold.
__global__ void cvt_adj_kernel(const float* __restrict__ adj,
                               __bf16* __restrict__ adj_bf,
                               float* __restrict__ rowsum) {
  int n = blockIdx.x & (NN - 1);
  int e = (blockIdx.x >> 9) & 7;
  int b = blockIdx.x >> 12;
  int t = threadIdx.x;                        // 128 threads, 4 elems each
  __bf16* dst = adj_bf + ((((long)b * EE + e) * NN + n) * NN);
  float s = 0.f;
  if (e < 4) {
    const float* src = adj + ((((long)b * 4 + e) * NN + n) * NN);
    float4 v = ((const float4*)src)[t];
    dst[t * 4 + 0] = (__bf16)v.x; dst[t * 4 + 1] = (__bf16)v.y;
    dst[t * 4 + 2] = (__bf16)v.z; dst[t * 4 + 3] = (__bf16)v.w;
    s = v.x + v.y + v.z + v.w;
  } else {
    const float* src = adj + (((long)b * 4 + (e - 4)) * NN * NN) + n; // column n
    #pragma unroll
    for (int j = 0; j < 4; ++j) {
      float x = src[(long)(t * 4 + j) * NN];
      dst[t * 4 + j] = (__bf16)x;
      s += x;
    }
  }
  __shared__ float red[128];
  red[t] = s;
  __syncthreads();
  for (int off = 64; off > 0; off >>= 1) {
    if (t < off) red[t] += red[t + off];
    __syncthreads();
  }
  if (t == 0) rowsum[((long)b * EE + e) * NN + n] = red[0];
}

// h0 = emb_table[labels] ; produce f32 (epilogue math) and bf16 (A operand).
__global__ void embed_kernel(const int* __restrict__ labels,
                             const float* __restrict__ table,
                             float* __restrict__ h, __bf16* __restrict__ hbf) {
  int bn = blockIdx.x;
  int d  = threadIdx.x;
  int lab = labels[bn];
  float v = table[(long)lab * DD + d];
  h  [(long)bn * DD + d] = v;
  hbf[(long)bn * DD + d] = (__bf16)v;
}

// Swizzle a (batched) f32 K x Nn row-major matrix into bf16 B-fragments.
// frag(kt,nt): lane holds col nt*16+(lane&15), K rows kt*32+(lane&16?8:0)+{0..7,16..23}.
__global__ void swz_b_kernel(const float* __restrict__ in, __bf16* __restrict__ out,
                             int K, int Nn, long in_bstride) {
  int lane = threadIdx.x;
  int ktiles = K >> 5, ntiles = Nn >> 4;
  int fpb = ktiles * ntiles;
  int f   = blockIdx.x % fpb;
  int bat = blockIdx.x / fpb;
  int kt = f / ntiles, nt = f % ntiles;
  const float* src = in + (long)bat * in_bstride;
  __bf16* dst = out + (((long)bat * fpb + f) << 9) + lane * 16;
  int col = nt * 16 + (lane & 15);
  int k0  = kt * 32 + ((lane & 16) ? 8 : 0);
  #pragma unroll
  for (int j = 0; j < 16; ++j)
    dst[j] = (__bf16)src[(long)(k0 + j + (j >= 8 ? 8 : 0)) * Nn + col];
}

// ---- per-step GEMMs ---------------------------------------------------------

// agg[b,e] = adj_bf[b,e] (NxN) @ h (NxD). One wave -> 16x64 tile (4 accs).
// Explicit double-buffer: next k-step's A + 4 B fragments are in flight while
// the current 4 WMMAs issue. K=512 -> 16 k-steps, 64 WMMA per wave.
__global__ void agg_kernel(const __bf16* __restrict__ adj_bf,
                           const __bf16* __restrict__ h_sw,
                           __bf16* __restrict__ agg) {
  int lane = threadIdx.x;
  int idx  = blockIdx.x;
  int dq = idx & 3, nt = (idx >> 2) & 31, e = (idx >> 7) & 7, b = idx >> 10;
  const __bf16* A   = adj_bf + (((long)b * EE + e) * NN * NN);
  const __bf16* Bsw = h_sw + (long)b * ((NN / 32) * (DD / 16) * 512);
  int row_base = nt * 16;
  v8f c[4] = {};
  v16bf af[2];
  v16bf bf[2][4];
  af[0] = load_a_frag(A, NN, row_base, 0, lane);
  #pragma unroll
  for (int j = 0; j < 4; ++j)
    bf[0][j] = load_b_frag(Bsw, dq * 4 + j, lane);
  #pragma unroll 4
  for (int kt = 0; kt < NN / 32; ++kt) {
    int cur = kt & 1, nxt = cur ^ 1;
    int kn  = (kt + 1) & (NN / 32 - 1);        // wraparound: no tail branch
    af[nxt] = load_a_frag(A, NN, row_base, kn * 32, lane);
    int fbn = kn * (DD / 16) + dq * 4;
    #pragma unroll
    for (int j = 0; j < 4; ++j)
      bf[nxt][j] = load_b_frag(Bsw, fbn + j, lane);
    __builtin_prefetch(A + (long)(row_base + (lane & 15)) * NN + ((kt + 2) & 15) * 32, 0, 1);
    #pragma unroll
    for (int j = 0; j < 4; ++j)
      c[j] = wmma_bf16(af[cur], bf[cur][j], c[j]);
  }
  __bf16* out = agg + (((long)b * EE + e) * NN * DD);
  int ro = (lane & 16) ? 8 : 0;
  #pragma unroll
  for (int j = 0; j < 4; ++j) {
    int col = (dq * 4 + j) * 16 + (lane & 15);
    #pragma unroll
    for (int r = 0; r < 8; ++r)
      out[(long)(row_base + r + ro) * DD + col] = (__bf16)c[j][r];
  }
}

// m[b] = sum_e agg[b,e] @ W_msg[e]  + rowsum[b,e,n]*b_msg[e,:]
// One wave -> 16x64 tile (4 accs), flat (e,kt) pipeline: 64 steps, 256 WMMA.
__global__ void msg_kernel(const __bf16* __restrict__ agg,
                           const __bf16* __restrict__ wmsg_sw,
                           const float* __restrict__ rowsum,
                           const float* __restrict__ b_msg,
                           __bf16* __restrict__ m_out) {
  int lane = threadIdx.x;
  int idx  = blockIdx.x;
  int dq = idx & 3, nt = (idx >> 2) & 31, b = idx >> 7;
  int row_base = nt * 16;
  const int KT = DD / 32;                      // 8 k-tiles per e
  v8f c[4] = {};
  v16bf af[2];
  v16bf bf[2][4];
  af[0] = load_a_frag(agg + ((long)b * EE) * NN * DD, DD, row_base, 0, lane);
  #pragma unroll
  for (int j = 0; j < 4; ++j)
    bf[0][j] = load_b_frag(wmsg_sw, dq * 4 + j, lane);
  #pragma unroll 2
  for (int i = 0; i < EE * KT; ++i) {
    int cur = i & 1, nxt = cur ^ 1;
    int in  = (i + 1) & (EE * KT - 1);
    int en = in / KT, ktn = in % KT;
    af[nxt] = load_a_frag(agg + (((long)b * EE + en) * NN * DD), DD,
                          row_base, ktn * 32, lane);
    const __bf16* Bswn = wmsg_sw + (long)en * (KT * (DD / 16) * 512);
    int fbn = ktn * (DD / 16) + dq * 4;
    #pragma unroll
    for (int j = 0; j < 4; ++j)
      bf[nxt][j] = load_b_frag(Bswn, fbn + j, lane);
    #pragma unroll
    for (int j = 0; j < 4; ++j)
      c[j] = wmma_bf16(af[cur], bf[cur][j], c[j]);
  }
  // bias fold: hoist b_msg per (e, col-tile)
  float bm[EE][4];
  #pragma unroll
  for (int e = 0; e < EE; ++e)
    #pragma unroll
    for (int j = 0; j < 4; ++j)
      bm[e][j] = b_msg[e * DD + (dq * 4 + j) * 16 + (lane & 15)];
  int ro = (lane & 16) ? 8 : 0;
  __bf16* out = m_out + (long)b * NN * DD;
  #pragma unroll
  for (int r = 0; r < 8; ++r) {
    int row = row_base + r + ro;
    float rs[EE];
    #pragma unroll
    for (int e = 0; e < EE; ++e)
      rs[e] = rowsum[((long)b * EE + e) * NN + row];
    #pragma unroll
    for (int j = 0; j < 4; ++j) {
      float v = c[j][r];
      #pragma unroll
      for (int e = 0; e < EE; ++e)
        v += rs[e] * bm[e][j];
      out[(long)row * DD + (dq * 4 + j) * 16 + (lane & 15)] = (__bf16)v;
    }
  }
}

// Fused GRU: gi = m @ w_ih, gh = h @ w_hh (3 gate tiles each), then gate math.
// All 6 B-fragments land in distinct registers before the 6 WMMAs issue.
__global__ void gru_kernel(const __bf16* __restrict__ m_bf,
                           const __bf16* __restrict__ h_bf,
                           const float* __restrict__ h_f32,
                           const __bf16* __restrict__ wih_sw,
                           const __bf16* __restrict__ whh_sw,
                           const float* __restrict__ b_ih,
                           const float* __restrict__ b_hh,
                           float* __restrict__ h_out,
                           __bf16* __restrict__ h_out_bf) {
  int lane = threadIdx.x;
  int idx  = blockIdx.x;
  int dt = idx & 15, nt = (idx >> 4) & 31, b = idx >> 9;
  int row_base = nt * 16;
  const __bf16* Am = m_bf + (long)b * NN * DD;
  const __bf16* Ah = h_bf + (long)b * NN * DD;
  v8f gi[3] = {};
  v8f gh[3] = {};
  const int NT3 = 3 * DD / 16;  // 48 col tiles in the 3D-wide weight
  #pragma unroll
  for (int kt = 0; kt < DD / 32; ++kt) {
    v16bf am = load_a_frag(Am, DD, row_base, kt * 32, lane);
    v16bf ah = load_a_frag(Ah, DD, row_base, kt * 32, lane);
    v16bf bi[3], bh[3];
    #pragma unroll
    for (int g = 0; g < 3; ++g) {
      bi[g] = load_b_frag(wih_sw, kt * NT3 + g * 16 + dt, lane);
      bh[g] = load_b_frag(whh_sw, kt * NT3 + g * 16 + dt, lane);
    }
    #pragma unroll
    for (int g = 0; g < 3; ++g) {
      gi[g] = wmma_bf16(am, bi[g], gi[g]);
      gh[g] = wmma_bf16(ah, bh[g], gh[g]);
    }
  }
  int col = dt * 16 + (lane & 15);
  float bir = b_ih[col], biz = b_ih[DD + col], bin = b_ih[2 * DD + col];
  float bhr = b_hh[col], bhz = b_hh[DD + col], bhn = b_hh[2 * DD + col];
  int ro = (lane & 16) ? 8 : 0;
  #pragma unroll
  for (int r = 0; r < 8; ++r) {
    int row = row_base + r + ro;
    float ir = gi[0][r] + bir, iz = gi[1][r] + biz, inn = gi[2][r] + bin;
    float hr = gh[0][r] + bhr, hz = gh[1][r] + bhz, hnn = gh[2][r] + bhn;
    float rg = 1.f / (1.f + __expf(-(ir + hr)));
    float zg = 1.f / (1.f + __expf(-(iz + hz)));
    float x  = inn + rg * hnn;
    float e2 = __expf(2.f * x);
    float ng = (e2 - 1.f) / (e2 + 1.f);          // tanh
    long o = ((long)b * NN + row) * DD + col;
    float hold = h_f32[o];
    float hnew = (1.f - zg) * ng + zg * hold;
    h_out[o]    = hnew;
    h_out_bf[o] = (__bf16)hnew;
  }
}

// ---- host launch ------------------------------------------------------------

extern "C" void kernel_launch(void* const* d_in, const int* in_sizes, int n_in,
                              void* d_out, int out_size, void* d_ws, size_t ws_size,
                              hipStream_t stream) {
  const float* adj    = (const float*)d_in[0];
  const int*   labels = (const int*)  d_in[1];
  // d_in[2] = n_prop_steps (device scalar; fixed to 8 by the problem; graph
  // capture forbids a sync readback, so the loop count is compiled in).
  const float* emb    = (const float*)d_in[3];
  const float* W_msg  = (const float*)d_in[4];
  const float* b_msg  = (const float*)d_in[5];
  const float* w_ih   = (const float*)d_in[6];
  const float* w_hh   = (const float*)d_in[7];
  const float* b_ih   = (const float*)d_in[8];
  const float* b_hh   = (const float*)d_in[9];

  char* ws = (char*)d_ws;
  size_t off = 0;
  auto alloc = [&](size_t bytes) { char* p = ws + off; off += (bytes + 255) & ~(size_t)255; return p; };

  __bf16* adj_bf  = (__bf16*)alloc((size_t)Bb * EE * NN * NN * 2);     // 134 MB (L2-resident)
  float*  rowsum  = (float*) alloc((size_t)Bb * EE * NN * 4);
  __bf16* wmsg_sw = (__bf16*)alloc((size_t)EE * (DD/32) * (DD/16) * 512 * 2);
  __bf16* wih_sw  = (__bf16*)alloc((size_t)(DD/32) * (3*DD/16) * 512 * 2);
  __bf16* whh_sw  = (__bf16*)alloc((size_t)(DD/32) * (3*DD/16) * 512 * 2);
  float*  hA      = (float*) alloc((size_t)Bb * NN * DD * 4);
  float*  hB      = (float*) alloc((size_t)Bb * NN * DD * 4);
  __bf16* hbfA    = (__bf16*)alloc((size_t)Bb * NN * DD * 2);
  __bf16* hbfB    = (__bf16*)alloc((size_t)Bb * NN * DD * 2);
  __bf16* hswA    = (__bf16*)alloc((size_t)Bb * (NN/32) * (DD/16) * 512 * 2);
  __bf16* hswB    = (__bf16*)alloc((size_t)Bb * (NN/32) * (DD/16) * 512 * 2);
  __bf16* agg_bf  = (__bf16*)alloc((size_t)Bb * EE * NN * DD * 2);
  __bf16* m_bf    = (__bf16*)alloc((size_t)Bb * NN * DD * 2);
  (void)ws_size; (void)in_sizes; (void)n_in; (void)out_size;

  // one-time prep
  cvt_adj_kernel<<<Bb * EE * NN, 128, 0, stream>>>(adj, adj_bf, rowsum);
  embed_kernel<<<Bb * NN, DD, 0, stream>>>(labels, emb, hA, hbfA);
  swz_b_kernel<<<EE * (DD/32) * (DD/16), 32, 0, stream>>>(W_msg, wmsg_sw, DD, DD, (long)DD * DD);
  swz_b_kernel<<<(DD/32) * (3*DD/16), 32, 0, stream>>>(w_ih, wih_sw, DD, 3 * DD, 0);
  swz_b_kernel<<<(DD/32) * (3*DD/16), 32, 0, stream>>>(w_hh, whh_sw, DD, 3 * DD, 0);
  swz_b_kernel<<<Bb * (NN/32) * (DD/16), 32, 0, stream>>>(hA, hswA, NN, DD, (long)NN * DD);

  float*  h_cur = hA;   float*  h_nxt = hB;
  __bf16* hb_cur = hbfA; __bf16* hb_nxt = hbfB;
  __bf16* hs_cur = hswA; __bf16* hs_nxt = hswB;

  for (int step = 0; step < STEPS; ++step) {
    agg_kernel<<<Bb * EE * (NN/16) * (DD/64), 32, 0, stream>>>(adj_bf, hs_cur, agg_bf);
    msg_kernel<<<Bb * (NN/16) * (DD/64), 32, 0, stream>>>(agg_bf, wmsg_sw, rowsum, b_msg, m_bf);
    float* hout = (step == STEPS - 1) ? (float*)d_out : h_nxt;
    gru_kernel<<<Bb * (NN/16) * (DD/16), 32, 0, stream>>>(
        m_bf, hb_cur, h_cur, wih_sw, whh_sw, b_ih, b_hh, hout, hb_nxt);
    if (step < STEPS - 1) {
      swz_b_kernel<<<Bb * (NN/32) * (DD/16), 32, 0, stream>>>(h_nxt, hs_nxt, NN, DD, (long)NN * DD);
      float* tf = h_cur; h_cur = h_nxt; h_nxt = tf;
      __bf16* tb = hb_cur; hb_cur = hb_nxt; hb_nxt = tb;
      __bf16* ts = hs_cur; hs_cur = hs_nxt; hs_nxt = ts;
    }
  }
}
